// GeometricGraphConv_73100343378529
// MI455X (gfx1250) — matrix operand order repaired and checked by reading, not compiled, not run on hardware
//
#include <hip/hip_runtime.h>
#include <hip/hip_fp16.h>

typedef __attribute__((ext_vector_type(16))) _Float16 v16h;
typedef __attribute__((ext_vector_type(8)))  float    v8f;
typedef __attribute__((ext_vector_type(4)))  int      v4i;
typedef _Float16 half_t;

#if defined(__has_builtin)
#if __has_builtin(__builtin_amdgcn_global_load_async_to_lds_b128) && \
    __has_builtin(__builtin_amdgcn_s_wait_asynccnt)
#define USE_ASYNC_GATHER 1
#endif
#endif
#ifndef USE_ASYNC_GATHER
#define USE_ASYNC_GATHER 0
#endif

union FragU { v16h h; uint4 q[2]; };

// A fragment (16x32 f16) from a row-major LDS tile.
// lane -> row = lane&15, hi = lane>>4; halfs j = K = k0 + 16*hi + j
__device__ __forceinline__ v16h lds_load_a(const half_t* base, int lda, int k0, int lane) {
  const int r = lane & 15, hi = lane >> 4;
  FragU f;
  f.q[0] = *(const uint4*)(base + r * lda + k0 + 16 * hi);
  f.q[1] = *(const uint4*)(base + r * lda + k0 + 16 * hi + 8);
  return f.h;
}

// B fragment (32x16 f16) from column-major f16 weights Wt[col*K + k].
// lane -> col = n0 + (lane&15); halfs j = K = k0 + 16*hi + j  (one contiguous 32B run)
__device__ __forceinline__ v16h gbl_load_b(const half_t* wt, int K, int n0, int k0, int lane) {
  const int c = n0 + (lane & 15), hi = lane >> 4;
  const uint4* p = (const uint4*)(wt + c * K + k0 + 16 * hi);
  FragU f;
  f.q[0] = p[0];
  f.q[1] = p[1];
  return f.h;
}

__device__ __forceinline__ v8f wmma16(v16h a, v16h b, v8f c) {
  return __builtin_amdgcn_wmma_f32_16x16x32_f16(false, a, false, b, (short)0, c, false, false);
}

// ---------------- weight prep: f32 row-major [K,64] -> f16 column-major [64,K] ----------------
__global__ void prep_weights(const float* Wn, const float* Wm1, const float* Wm2,
                             const float* Wu1, const float* Wu2,
                             half_t* Wn_t, half_t* Wm1_t, float* Wm1_last,
                             half_t* Wm2_t, half_t* Wu1_t, half_t* Wu2_t) {
  int g = blockIdx.x * blockDim.x + threadIdx.x;
  if (g < 4096) {                    // three 64x64 matrices (K=64)
    int c = g >> 6, k = g & 63;
    Wn_t [c * 64 + k] = (half_t)Wn [k * 64 + c];
    Wm2_t[c * 64 + k] = (half_t)Wm2[k * 64 + c];
    Wu2_t[c * 64 + k] = (half_t)Wu2[k * 64 + c];
  } else if (g < 4096 + 8192) {      // two 128x64 matrices (K=128); Wm1 uses first 128 rows
    int q = g - 4096;
    int c = q >> 7, k = q & 127;
    Wm1_t[c * 128 + k] = (half_t)Wm1[k * 64 + c];
    Wu1_t[c * 128 + k] = (half_t)Wu1[k * 64 + c];
  } else if (g < 4096 + 8192 + 64) { // edge_attr column of W_m1 (row 128), kept f32
    int c = g - 4096 - 8192;
    Wm1_last[c] = Wm1[128 * 64 + c];
  }
}

// ---------------- node GEMM: h_pre = x @ W_node + b (16 rows/block, 4 waves x 16 cols) -------
__global__ void node_gemm(const float* x, const half_t* Wn_t, const float* bn,
                          float* h_pre, int n_nodes) {
  __shared__ half_t Af[16 * 64];
  const int tile = blockIdx.x, t = threadIdx.x;
  const bool full = (tile * 16 + 16 <= n_nodes);
  if (full) {
    for (int idx = t; idx < 16 * 64; idx += 128) {
      int r = idx >> 6, c = idx & 63;
      Af[idx] = (half_t)x[(tile * 16 + r) * 64 + c];
    }
  } else {
    for (int idx = t; idx < 16 * 64; idx += 128) {
      int r = idx >> 6, c = idx & 63;
      int node = tile * 16 + r;
      Af[idx] = (node < n_nodes) ? (half_t)x[node * 64 + c] : (half_t)0.0f;
    }
  }
  __syncthreads();
  const int w = t >> 5, lane = t & 31, n0 = w * 16;
  v8f acc = {};
  for (int k0 = 0; k0 < 64; k0 += 32) {
    v16h a = lds_load_a(Af, 64, k0, lane);
    v16h b = gbl_load_b(Wn_t, 64, n0, k0, lane);
    acc = wmma16(a, b, acc);
  }
  const int col = n0 + (lane & 15), hi = lane >> 4;
  const float bias = bn[col];
  if (full) {
    for (int v = 0; v < 8; ++v)
      h_pre[(tile * 16 + v + 8 * hi) * 64 + col] = acc[v] + bias;
  } else {
    for (int v = 0; v < 8; ++v) {
      int node = tile * 16 + v + 8 * hi;
      if (node < n_nodes) h_pre[node * 64 + col] = acc[v] + bias;
    }
  }
}

// ---------------- BN training-mode column stats -> affine scale/shift -----------------------
__global__ void bn_stats(const float* h_pre, const float* gamma, const float* beta,
                         float* scale, float* shift, int n_nodes) {
  __shared__ float s1[256], s2[256];
  const int c = blockIdx.x, t = threadIdx.x;
  float sum = 0.f, sq = 0.f;
  for (int r = t; r < n_nodes; r += 256) {
    float v = h_pre[r * 64 + c];
    sum += v; sq += v * v;
  }
  s1[t] = sum; s2[t] = sq;
  __syncthreads();
  for (int off = 128; off > 0; off >>= 1) {
    if (t < off) { s1[t] += s1[t + off]; s2[t] += s2[t + off]; }
    __syncthreads();
  }
  if (t == 0) {
    float mean = s1[0] / n_nodes;
    float var  = s2[0] / n_nodes - mean * mean;   // biased, as torch BN
    float sc = gamma[c] * rsqrtf(var + 1e-5f);
    scale[c] = sc;
    shift[c] = beta[c] - mean * sc;
  }
}

__global__ void bn_apply(const float* h_pre, const float* scale, const float* shift,
                         half_t* h16, int total) {
  for (int i = blockIdx.x * blockDim.x + threadIdx.x; i < total; i += gridDim.x * blockDim.x) {
    int c = i & 63;
    float v = h_pre[i] * scale[c] + shift[c];
    h16[i] = (half_t)fmaxf(v, 0.f);
  }
}

__global__ void zero_f32(float* p, long long n) {
  for (long long i = (long long)blockIdx.x * blockDim.x + threadIdx.x; i < n;
       i += (long long)gridDim.x * blockDim.x)
    p[i] = 0.f;
}

// ---------------- edge messages: gather -> WMMA -> ReLU -> WMMA -> atomic scatter -----------
__global__ void edge_kernel(const half_t* h16, const long long* ei, const float* eattr,
                            const half_t* Wm1_t, const float* Wm1_last, const float* bm1,
                            const half_t* Wm2_t, const float* bm2,
                            float* s, float* cnt, int n_edges) {
  __shared__ long long srcI[16], dstI[16];
  __shared__ float ea[16];
  __shared__ half_t Af[16 * 128];   // [h[dst] | h[src]] per edge row
  __shared__ half_t M1[16 * 64];    // relu(msg_in @ Wm1 + b), f16
  const int tile = blockIdx.x, t = threadIdx.x;
  const bool full = (tile * 16 + 16 <= n_edges);
  if (t < 16) {
    long long e = (long long)tile * 16 + t;
    if (e < n_edges) {
      srcI[t] = ei[e];                         // edge_index[0] = source j
      dstI[t] = ei[(long long)n_edges + e];    // edge_index[1] = target i
      ea[t]   = eattr[e];
    } else { srcI[t] = -1; dstI[t] = -1; ea[t] = 0.f; }
  }
  const int w = t >> 5, lane = t & 31, n0 = w * 16;
  // preload all weight fragments before the gather barrier (overlap with staging)
  v16h b10 = gbl_load_b(Wm1_t, 128, n0, 0,  lane);
  v16h b11 = gbl_load_b(Wm1_t, 128, n0, 32, lane);
  v16h b12 = gbl_load_b(Wm1_t, 128, n0, 64, lane);
  v16h b13 = gbl_load_b(Wm1_t, 128, n0, 96, lane);
  v16h b20 = gbl_load_b(Wm2_t, 64,  n0, 0,  lane);
  v16h b21 = gbl_load_b(Wm2_t, 64,  n0, 32, lane);
  __syncthreads();
#if USE_ASYNC_GATHER
  if (full) {
    // CDNA5 async global->LDS gather: 256 x 16B, ASYNCcnt-tracked, no VGPR staging
    for (int q = t; q < 256; q += 128) {
      int r = q >> 4, part = q & 15;
      long long node = (part < 8) ? dstI[r] : srcI[r];
      __builtin_amdgcn_global_load_async_to_lds_b128(
          (__attribute__((address_space(1))) v4i*)(h16 + node * 64 + (part & 7) * 8),
          (__attribute__((address_space(3))) v4i*)(Af + r * 128 + part * 8),
          0, 0);
    }
    __builtin_amdgcn_s_wait_asynccnt(0);
    __syncthreads();
  } else
#endif
  {
    for (int q = t; q < 256; q += 128) {
      int r = q >> 4, part = q & 15;
      uint4* dp = (uint4*)(Af + r * 128 + part * 8);
      long long node = (part < 8) ? dstI[r] : srcI[r];
      if (node >= 0) *dp = *(const uint4*)(h16 + node * 64 + (part & 7) * 8);
      else           { uint4 z = {0, 0, 0, 0}; *dp = z; }
    }
    __syncthreads();
  }
  const int ci = lane & 15, hi = lane >> 4, col = n0 + ci;
  // GEMM1: [16x128] @ [128x16] + edge_attr rank-1 + bias, ReLU
  v8f acc = {};
  acc = wmma16(lds_load_a(Af, 128, 0,  lane), b10, acc);
  acc = wmma16(lds_load_a(Af, 128, 32, lane), b11, acc);
  acc = wmma16(lds_load_a(Af, 128, 64, lane), b12, acc);
  acc = wmma16(lds_load_a(Af, 128, 96, lane), b13, acc);
  const float wl = Wm1_last[col], b1 = bm1[col];
  for (int v = 0; v < 8; ++v) {
    int r = v + 8 * hi;
    M1[r * 64 + col] = (half_t)fmaxf(acc[v] + ea[r] * wl + b1, 0.f);
  }
  __syncthreads();
  // GEMM2: [16x64] @ [64x16] + bias, scatter-add into s[dst]
  v8f acc2 = {};
  acc2 = wmma16(lds_load_a(M1, 64, 0,  lane), b20, acc2);
  acc2 = wmma16(lds_load_a(M1, 64, 32, lane), b21, acc2);
  const float b2 = bm2[col];
  if (full) {
    for (int v = 0; v < 8; ++v)
      atomicAdd(&s[dstI[v + 8 * hi] * 64 + col], acc2[v] + b2);
    if (t < 16) atomicAdd(&cnt[dstI[t]], 1.0f);
  } else {
    for (int v = 0; v < 8; ++v) {
      long long node = dstI[v + 8 * hi];
      if (node >= 0) atomicAdd(&s[node * 64 + col], acc2[v] + b2);
    }
    if (t < 16 && dstI[t] >= 0) atomicAdd(&cnt[dstI[t]], 1.0f);
  }
}

// ---------------- update MLP: out = ReLU([h|agg] @ Wu1 + b) @ Wu2 + b -----------------------
__global__ void update_kernel(const half_t* h16, const float* s, const float* cnt,
                              const half_t* Wu1_t, const float* bu1,
                              const half_t* Wu2_t, const float* bu2,
                              float* out, int n_nodes) {
  __shared__ half_t Af[16 * 128];
  __shared__ half_t M1[16 * 64];
  __shared__ float inv[16];
  const int tile = blockIdx.x, t = threadIdx.x;
  const bool full = (tile * 16 + 16 <= n_nodes);
  if (t < 16) {
    int node = tile * 16 + t;
    inv[t] = (node < n_nodes) ? 1.0f / fmaxf(cnt[node], 1.0f) : 0.f;
  }
  const int w = t >> 5, lane = t & 31, n0 = w * 16;
  v16h b10 = gbl_load_b(Wu1_t, 128, n0, 0,  lane);
  v16h b11 = gbl_load_b(Wu1_t, 128, n0, 32, lane);
  v16h b12 = gbl_load_b(Wu1_t, 128, n0, 64, lane);
  v16h b13 = gbl_load_b(Wu1_t, 128, n0, 96, lane);
  v16h b20 = gbl_load_b(Wu2_t, 64,  n0, 0,  lane);
  v16h b21 = gbl_load_b(Wu2_t, 64,  n0, 32, lane);
  __syncthreads();
  if (full) {
    for (int idx = t; idx < 16 * 64; idx += 128) {
      int r = idx >> 6, c = idx & 63;
      int node = tile * 16 + r;
      Af[r * 128 + c]      = h16[node * 64 + c];
      Af[r * 128 + 64 + c] = (half_t)(s[node * 64 + c] * inv[r]);   // scatter-mean
    }
  } else {
    for (int idx = t; idx < 16 * 64; idx += 128) {
      int r = idx >> 6, c = idx & 63;
      int node = tile * 16 + r;
      if (node < n_nodes) {
        Af[r * 128 + c]      = h16[node * 64 + c];
        Af[r * 128 + 64 + c] = (half_t)(s[node * 64 + c] * inv[r]);
      } else {
        Af[r * 128 + c] = (half_t)0.f;
        Af[r * 128 + 64 + c] = (half_t)0.f;
      }
    }
  }
  __syncthreads();
  const int ci = lane & 15, hi = lane >> 4, col = n0 + ci;
  v8f acc = {};
  acc = wmma16(lds_load_a(Af, 128, 0,  lane), b10, acc);
  acc = wmma16(lds_load_a(Af, 128, 32, lane), b11, acc);
  acc = wmma16(lds_load_a(Af, 128, 64, lane), b12, acc);
  acc = wmma16(lds_load_a(Af, 128, 96, lane), b13, acc);
  const float b1 = bu1[col];
  for (int v = 0; v < 8; ++v) {
    int r = v + 8 * hi;
    M1[r * 64 + col] = (half_t)fmaxf(acc[v] + b1, 0.f);
  }
  __syncthreads();
  v8f acc2 = {};
  acc2 = wmma16(lds_load_a(M1, 64, 0,  lane), b20, acc2);
  acc2 = wmma16(lds_load_a(M1, 64, 32, lane), b21, acc2);
  const float b2 = bu2[col];
  if (full) {
    for (int v = 0; v < 8; ++v)
      out[(tile * 16 + v + 8 * hi) * 64 + col] = acc2[v] + b2;
  } else {
    for (int v = 0; v < 8; ++v) {
      int node = tile * 16 + v + 8 * hi;
      if (node < n_nodes) out[node * 64 + col] = acc2[v] + b2;
    }
  }
}

// ---------------- host launcher --------------------------------------------------------------
extern "C" void kernel_launch(void* const* d_in, const int* in_sizes, int n_in,
                              void* d_out, int out_size, void* d_ws, size_t ws_size,
                              hipStream_t stream) {
  const float*      x      = (const float*)d_in[0];
  const long long*  eidx   = (const long long*)d_in[1];   // int64 [2,E]
  const float*      eattr  = (const float*)d_in[2];       // [E,1]
  const float*      Wn     = (const float*)d_in[3];
  const float*      bnode  = (const float*)d_in[4];
  const float*      gamma  = (const float*)d_in[5];
  const float*      beta   = (const float*)d_in[6];
  const float*      Wm1    = (const float*)d_in[7];
  const float*      bm1    = (const float*)d_in[8];
  const float*      Wm2    = (const float*)d_in[9];
  const float*      bm2    = (const float*)d_in[10];
  const float*      Wu1    = (const float*)d_in[11];
  const float*      bu1    = (const float*)d_in[12];
  const float*      Wu2    = (const float*)d_in[13];
  const float*      bu2    = (const float*)d_in[14];
  float*            outp   = (float*)d_out;

  const int Nn = in_sizes[0] / 64;      // 50000
  const int Ee = in_sizes[2];           // 800000

  // workspace layout (256B aligned bumps); h_pre region is reused as scatter accumulator s
  char* wsp = (char*)d_ws;
  auto bump = [&](size_t bytes) -> void* {
    void* p = wsp; wsp += (bytes + 255) & ~(size_t)255; return p;
  };
  float*  h_pre    = (float*)bump((size_t)Nn * 64 * 4);   // later: s
  half_t* h16      = (half_t*)bump((size_t)Nn * 64 * 2);
  float*  cnt      = (float*)bump((size_t)Nn * 4);
  float*  scale    = (float*)bump(64 * 4);
  float*  shift    = (float*)bump(64 * 4);
  half_t* Wn_t     = (half_t*)bump(4096 * 2);
  half_t* Wm1_t    = (half_t*)bump(8192 * 2);
  float*  Wm1_last = (float*)bump(64 * 4);
  half_t* Wm2_t    = (half_t*)bump(4096 * 2);
  half_t* Wu1_t    = (half_t*)bump(8192 * 2);
  half_t* Wu2_t    = (half_t*)bump(4096 * 2);
  float*  s        = h_pre;             // alias: h_pre dead after bn_apply

  const int nodeTiles = (Nn + 15) / 16;
  const int edgeTiles = (Ee + 15) / 16;

  prep_weights<<<(4096 + 8192 + 64 + 255) / 256, 256, 0, stream>>>(
      Wn, Wm1, Wm2, Wu1, Wu2, Wn_t, Wm1_t, Wm1_last, Wm2_t, Wu1_t, Wu2_t);

  node_gemm<<<nodeTiles, 128, 0, stream>>>(x, Wn_t, bnode, h_pre, Nn);

  bn_stats<<<64, 256, 0, stream>>>(h_pre, gamma, beta, scale, shift, Nn);
  bn_apply<<<512, 256, 0, stream>>>(h_pre, scale, shift, h16, Nn * 64);

  zero_f32<<<512, 256, 0, stream>>>(s, (long long)Nn * 64);
  zero_f32<<<64, 256, 0, stream>>>(cnt, (long long)Nn);

  edge_kernel<<<edgeTiles, 128, 0, stream>>>(h16, eidx, eattr, Wm1_t, Wm1_last, bm1,
                                             Wm2_t, bm2, s, cnt, Ee);

  update_kernel<<<nodeTiles, 128, 0, stream>>>(h16, s, cnt, Wu1_t, bu1, Wu2_t, bu2,
                                               outp, Nn);
}
